// LinearAttention_55018531061832
// MI455X (gfx1250) — matrix operand order repaired and checked by reading, not compiled, not run on hardware
//
#include <hip/hip_runtime.h>

#define D_MODEL 1024
#define NHEADS  16
#define DK      64
#define BATCH   4
#define SEQ     4096
#define MROWS   (BATCH * SEQ)   // 16384

typedef __attribute__((ext_vector_type(16))) __bf16 v16bf;
typedef __attribute__((ext_vector_type(8)))  __bf16 v8bf;
typedef __attribute__((ext_vector_type(4)))  __bf16 v4bf;
typedef __attribute__((ext_vector_type(8)))  float  v8f;
typedef __attribute__((ext_vector_type(4)))  unsigned int v4u;
typedef __attribute__((ext_vector_type(8)))  int    v8i;
typedef __attribute__((ext_vector_type(4)))  int    v4i;

// ---------------------------------------------------------------------------
// WMMA fragment loaders (wave32 layouts per CDNA5 ISA §7.12.2)
// ---------------------------------------------------------------------------
// A-matrix fragment (16x32 bf16, M x K). Source rows K-contiguous.
// Lane L: m = L%16, half = L/16. VGPR0-3: K = half*8+0..7 ; VGPR4-7: K = 16+half*8+0..7.
__device__ __forceinline__ v16bf fragA_ld(const __bf16* base, int stride, int lane) {
    const int m = lane & 15, hv = lane >> 4;
    const __bf16* p = base + (size_t)m * stride + hv * 8;
    v16bf r;
    *(v8bf*)&r         = *(const v8bf*)(p);
    *(((v8bf*)&r) + 1) = *(const v8bf*)(p + 16);
    return r;
}

// B-matrix fragment (32x16 bf16, K x N). Source: B^T rows (n-major, K contiguous).
// Lane L: n = L%16, half = L/16. VGPR0-3: K = half*8+0..7 ; VGPR4-7: K = 16+half*8+0..7.
__device__ __forceinline__ v16bf fragB_ld(const __bf16* base, int stride, int lane) {
    const int n = lane & 15, hv = lane >> 4;
    const __bf16* p = base + (size_t)n * stride + hv * 8;
    v16bf r;
    *(v8bf*)&r         = *(const v8bf*)(p);
    *(((v8bf*)&r) + 1) = *(const v8bf*)(p + 16);
    return r;
}

// ---------------------------------------------------------------------------
// Tensor Data Mover: 2D tile (bf16) global -> LDS. D# per ISA 08_async_tensor §8.
// group0: [count=1][lds_addr][global_addr 56:0 | type=2]
// group1: [wg_mask=0|data_size=1(2B)] [tensor_dim0] [tensor_dim1] [tile_dim0]
//         [tile_dim1] [tensor_dim0_stride(48b)] [tensor_dim1_stride=0]
// 6-arg builtin (clang-23 / therock form): (v4u, v8i, v4i, v4i, v8i, i32 cpol)
// ---------------------------------------------------------------------------
__device__ __forceinline__ void tdm_load_2d(const __bf16* gsrc, unsigned lds_byte_off,
                                            int tile_rows, int tile_cols,
                                            int tensor_rows, int tensor_cols) {
    const unsigned long long ga = (unsigned long long)(const void*)gsrc;
    v4u g0;
    g0[0] = 1u;                                        // count=1, user mode
    g0[1] = lds_byte_off;                              // lds_addr
    g0[2] = (unsigned)(ga & 0xffffffffu);              // global_addr[31:0]
    g0[3] = (unsigned)((ga >> 32) & 0x01ffffffu)       // global_addr[56:32]
          | 0x80000000u;                               // type = 2 ("image")
    v8i g1;
    g1[0] = (int)(1u << 16);                           // wg_mask=0, data_size=1 (2B)
    g1[1] = (int)(((unsigned)tensor_cols & 0xffffu) << 16);          // tensor_dim0 lo
    g1[2] = (int)((((unsigned)tensor_cols >> 16) & 0xffffu)
          | (((unsigned)tensor_rows & 0xffffu) << 16));              // dim0 hi | dim1 lo
    g1[3] = (int)((((unsigned)tensor_rows >> 16) & 0xffffu)
          | (((unsigned)tile_cols & 0xffffu) << 16));                // dim1 hi | tile_dim0
    g1[4] = (int)((unsigned)tile_rows & 0xffffu);                    // tile_dim1, tile_dim2=0
    g1[5] = (int)(unsigned)tensor_cols;                              // tensor_dim0_stride lo32
    g1[6] = 0;                                                       // stride hi | dim1_stride lo
    g1[7] = 0;
    const v4i gz4 = {0, 0, 0, 0};                      // groups 2/3 unused (2D tensor)
    const v8i gz8 = {0, 0, 0, 0, 0, 0, 0, 0};          // extra descriptor words (unused)
    __builtin_amdgcn_tensor_load_to_lds(g0, g1, gz4, gz4, gz8, 0);
}

__device__ __forceinline__ unsigned lds_off(const void* p) {
    // generic LDS pointer = {SHARED aperture, offset}; low 32 bits = LDS byte offset
    return (unsigned)(unsigned long long)p;
}

// ---------------------------------------------------------------------------
// GEMM: out = X @ W^T + bias (X, W pre-converted bf16); optional elu+1.
// STORE: 0 bf16 [M,1024] | 1 bf16 per-head transposed [b][h][d][s] | 2 f32 [M,1024]
// Block 256 = 8 waves; tile 128(M) x 64(N); K-step 64; TDM double-buffered LDS.
// ---------------------------------------------------------------------------
template <int ACT, int STORE>
__global__ __launch_bounds__(256) void gemm_bf16(const __bf16* __restrict__ X,
                                                 const __bf16* __restrict__ Wb,
                                                 const float* __restrict__ bias,
                                                 void* __restrict__ out) {
    const int tid  = threadIdx.x;
    const int lane = tid & 31, wave = tid >> 5;
    const int wm = wave & 3;   // 4 waves along M (32 rows)
    const int wn = wave >> 2;  // 2 waves along N (32 cols)
    const int m0 = blockIdx.y * 128;
    const int n0 = blockIdx.x * 64;

    __shared__ __bf16 sA[2][128 * 64];
    __shared__ __bf16 sB[2][64 * 64];

    const __bf16* atile = X  + (size_t)m0 * D_MODEL;
    const __bf16* btile = Wb + (size_t)n0 * D_MODEL;

    // prologue: stage 0 via TDM (one wave issues; TDM ignores EXEC, 1 issue/wave)
    if (wave == 0) {
        tdm_load_2d(atile, lds_off(&sA[0][0]), 128, 64, MROWS,   D_MODEL);
        tdm_load_2d(btile, lds_off(&sB[0][0]), 64,  64, D_MODEL, D_MODEL);
    }

    v8f acc[2][2] = {};

#pragma unroll 1
    for (int kt = 0; kt < D_MODEL / 64; ++kt) {
        if (wave == 0) {
            if (kt + 1 < D_MODEL / 64) {
                const int nb = (kt + 1) & 1;
                tdm_load_2d(atile + (kt + 1) * 64, lds_off(&sA[nb][0]), 128, 64, MROWS,   D_MODEL);
                tdm_load_2d(btile + (kt + 1) * 64, lds_off(&sB[nb][0]), 64,  64, D_MODEL, D_MODEL);
                __builtin_amdgcn_s_wait_tensorcnt(2);  // in-order: stage kt complete
            } else {
                __builtin_amdgcn_s_wait_tensorcnt(0);
            }
        }
        __syncthreads();

        const __bf16* A0 = &sA[kt & 1][0];
        const __bf16* B0 = &sB[kt & 1][0];
#pragma unroll
        for (int kc = 0; kc < 2; ++kc) {
            v16bf a[2], b[2];
#pragma unroll
            for (int mt = 0; mt < 2; ++mt)
                a[mt] = fragA_ld(A0 + (wm * 32 + mt * 16) * 64 + kc * 32, 64, lane);
#pragma unroll
            for (int nt = 0; nt < 2; ++nt)
                b[nt] = fragB_ld(B0 + (wn * 32 + nt * 16) * 64 + kc * 32, 64, lane);
#pragma unroll
            for (int mt = 0; mt < 2; ++mt)
#pragma unroll
                for (int nt = 0; nt < 2; ++nt)
                    acc[mt][nt] = __builtin_amdgcn_wmma_f32_16x16x32_bf16(
                        false, a[mt], false, b[nt], (short)0, acc[mt][nt], false, false);
        }
        __syncthreads();
    }

    // epilogue: bias, activation, store
    const int hv = lane >> 4, ln = lane & 15;
#pragma unroll
    for (int mt = 0; mt < 2; ++mt) {
#pragma unroll
        for (int nt = 0; nt < 2; ++nt) {
            const int n_g = n0 + wn * 32 + nt * 16 + ln;
            const float bv = bias[n_g];
#pragma unroll
            for (int r = 0; r < 8; ++r) {
                const int m_g = m0 + wm * 32 + mt * 16 + hv * 8 + r;
                float v = acc[mt][nt][r] + bv;
                if constexpr (ACT == 1) v = (v > 0.f) ? (v + 1.f) : __expf(v);
                if constexpr (STORE == 0) {
                    ((__bf16*)out)[(size_t)m_g * D_MODEL + n_g] = (__bf16)v;
                } else if constexpr (STORE == 1) {
                    const int b_ = m_g >> 12, s_ = m_g & (SEQ - 1);
                    const int h_ = n_g >> 6,  d_ = n_g & (DK - 1);
                    ((__bf16*)out)[(size_t)((b_ * NHEADS + h_) * DK + d_) * SEQ + s_] = (__bf16)v;
                } else {
                    ((float*)out)[(size_t)m_g * D_MODEL + n_g] = v;
                }
            }
        }
    }
}

// ---------------------------------------------------------------------------
// KV^T[e][d] = sum_s V[s][e] * K[s][d] per (b,h); A-frags from Vt, B-frags from
// Kt (both [bh][d][s]); pure global b128 loads; f32 atomic accumulate.
// ---------------------------------------------------------------------------
__global__ __launch_bounds__(128) void kv_accum(const __bf16* __restrict__ Kt,
                                                const __bf16* __restrict__ Vt,
                                                float* __restrict__ KVf) {
    const int lane = threadIdx.x & 31, wave = threadIdx.x >> 5;
    const int s0 = blockIdx.x * 512;
    const int bh = blockIdx.y;
    const __bf16* vb = Vt + ((size_t)bh * DK + wave * 16) * SEQ + s0;
    const __bf16* kb = Kt + (size_t)bh * DK * SEQ + s0;

    v8f acc[4] = {};
    for (int it = 0; it < 16; ++it) {
        const int ks = it * 32;
        const v16bf a = fragA_ld(vb + ks, SEQ, lane);
#pragma unroll
        for (int nt = 0; nt < 4; ++nt) {
            const v16bf b = fragB_ld(kb + (size_t)(nt * 16) * SEQ + ks, SEQ, lane);
            acc[nt] = __builtin_amdgcn_wmma_f32_16x16x32_bf16(
                false, a, false, b, (short)0, acc[nt], false, false);
        }
    }
    const int hv = lane >> 4, ln = lane & 15;
    float* outb = KVf + (size_t)bh * (DK * DK);
#pragma unroll
    for (int nt = 0; nt < 4; ++nt)
#pragma unroll
        for (int r = 0; r < 8; ++r) {
            const int e = wave * 16 + hv * 8 + r;
            const int d = nt * 16 + ln;
            atomicAdd(&outb[e * DK + d], acc[nt][r]);
        }
}

// Ksum[bh*64 + d] = sum_s Kt[bh][d][s]; one wave per row.
__global__ __launch_bounds__(256) void ksum_kernel(const __bf16* __restrict__ Kt,
                                                   __bf16* __restrict__ Ksum) {
    const int gw = (int)((blockIdx.x * blockDim.x + threadIdx.x) >> 5);
    const int lane = threadIdx.x & 31;
    const __bf16* row = Kt + (size_t)gw * SEQ;
    float s = 0.f;
    for (int i = lane; i < SEQ; i += 32) s += (float)row[i];
#pragma unroll
    for (int off = 16; off; off >>= 1) s += __shfl_xor(s, off, 32);
    if (lane == 0) Ksum[gw] = (__bf16)s;
}

__global__ void zero_f32(float* p, int n) {
    const int i = blockIdx.x * blockDim.x + threadIdx.x;
    if (i < n) p[i] = 0.f;
}

// f32 -> bf16, 4 elements per thread
__global__ void cvt_f32_bf16(const float* __restrict__ in, __bf16* __restrict__ out, int n4) {
    const int i = blockIdx.x * blockDim.x + threadIdx.x;
    if (i < n4) {
        const float4 f = *(const float4*)(in + (size_t)i * 4);
        v4bf v;
        v[0] = (__bf16)f.x; v[1] = (__bf16)f.y; v[2] = (__bf16)f.z; v[3] = (__bf16)f.w;
        *(v4bf*)(out + (size_t)i * 4) = v;
    }
}

// ---------------------------------------------------------------------------
// Z[s][e] = (Q[s] . KV[:,e]) / (Q[s] . Ksum + 1e-6); WMMA + shuffle normalizer.
// ---------------------------------------------------------------------------
__global__ __launch_bounds__(128) void attn_apply(const __bf16* __restrict__ Q,
                                                  const __bf16* __restrict__ KVt,
                                                  const __bf16* __restrict__ Ksum,
                                                  __bf16* __restrict__ Z) {
    const int lane = threadIdx.x & 31, wave = threadIdx.x >> 5;
    const int bh = blockIdx.y;
    const int b_ = bh >> 4, h_ = bh & 15;
    const int s0 = blockIdx.x * 64 + wave * 16;

    const __bf16* qbase = Q + ((size_t)b_ * SEQ + s0) * D_MODEL + h_ * DK;
    const __bf16* kvb   = KVt + (size_t)bh * (DK * DK);
    const __bf16* ksb   = Ksum + (size_t)bh * DK;

    v16bf aq[2];
#pragma unroll
    for (int kc = 0; kc < 2; ++kc) aq[kc] = fragA_ld(qbase + kc * 32, D_MODEL, lane);

    v8f acc[4] = {};
#pragma unroll
    for (int kc = 0; kc < 2; ++kc)
#pragma unroll
        for (int nt = 0; nt < 4; ++nt) {
            const v16bf bkv = fragB_ld(kvb + (size_t)(nt * 16) * DK + kc * 32, DK, lane);
            acc[nt] = __builtin_amdgcn_wmma_f32_16x16x32_bf16(
                false, aq[kc], false, bkv, (short)0, acc[nt], false, false);
        }

    // D = Q . Ksum from the already-loaded A-fragments
    const int hv = lane >> 4;
    float part = 0.f;
#pragma unroll
    for (int kc = 0; kc < 2; ++kc) {
        const __bf16* p = ksb + kc * 32 + hv * 8;
#pragma unroll
        for (int i = 0; i < 8; ++i) {
            part += (float)aq[kc][i]     * (float)p[i];
            part += (float)aq[kc][i + 8] * (float)p[16 + i];
        }
    }
    const float Dm = part + __shfl_xor(part, 16, 32);  // full D for row (lane&15)

    __bf16* zb = Z + ((size_t)b_ * SEQ + s0) * D_MODEL + h_ * DK;
#pragma unroll
    for (int r = 0; r < 8; ++r) {
        const float Dr  = __shfl(Dm, hv * 8 + r, 32);
        const float inv = 1.f / (Dr + 1e-6f);
        const int m = hv * 8 + r;
#pragma unroll
        for (int nt = 0; nt < 4; ++nt)
            zb[(size_t)m * D_MODEL + nt * 16 + (lane & 15)] = (__bf16)(acc[nt][r] * inv);
    }
}

extern "C" void kernel_launch(void* const* d_in, const int* in_sizes, int n_in,
                              void* d_out, int out_size, void* d_ws, size_t ws_size,
                              hipStream_t stream) {
    (void)in_sizes; (void)n_in; (void)out_size; (void)ws_size;
    const float* query = (const float*)d_in[0];
    const float* key   = (const float*)d_in[1];
    const float* value = (const float*)d_in[2];
    const float* w_q = (const float*)d_in[3];
    const float* b_q = (const float*)d_in[4];
    const float* w_k = (const float*)d_in[5];
    const float* b_k = (const float*)d_in[6];
    const float* w_v = (const float*)d_in[7];
    const float* b_v = (const float*)d_in[8];
    const float* w_o = (const float*)d_in[9];
    const float* b_o = (const float*)d_in[10];

    char* ws = (char*)d_ws;
    __bf16* Qbuf  = (__bf16*)ws; ws += (size_t)MROWS * D_MODEL * 2;
    __bf16* Ktbuf = (__bf16*)ws; ws += (size_t)MROWS * D_MODEL * 2;
    __bf16* Vtbuf = (__bf16*)ws; ws += (size_t)MROWS * D_MODEL * 2;
    __bf16* Zbuf  = (__bf16*)ws; ws += (size_t)MROWS * D_MODEL * 2;
    float*  KVf   = (float*)ws;  ws += (size_t)BATCH * NHEADS * DK * DK * 4;
    __bf16* KVtb  = (__bf16*)ws; ws += (size_t)BATCH * NHEADS * DK * DK * 2;
    __bf16* Ksumb = (__bf16*)ws; ws += (size_t)BATCH * NHEADS * DK * 2 + 64;
    __bf16* Xb    = (__bf16*)ws; ws += (size_t)MROWS * D_MODEL * 2;       // staged bf16 activations
    __bf16* Wbuf  = (__bf16*)ws; ws += (size_t)D_MODEL * D_MODEL * 2;     // staged bf16 weights

    const int nx4 = MROWS * D_MODEL / 4;     // 4.19M float4s
    const int nw4 = D_MODEL * D_MODEL / 4;   // 262144
    const dim3 gg(D_MODEL / 64, MROWS / 128);  // (16, 128)

    // Q projection (elu+1, row-major bf16)
    cvt_f32_bf16<<<(nx4 + 255) / 256, 256, 0, stream>>>(query, Xb, nx4);
    cvt_f32_bf16<<<(nw4 + 255) / 256, 256, 0, stream>>>(w_q, Wbuf, nw4);
    gemm_bf16<1, 0><<<gg, 256, 0, stream>>>(Xb, Wbuf, b_q, Qbuf);
    // K projection (elu+1, per-head transposed)
    cvt_f32_bf16<<<(nx4 + 255) / 256, 256, 0, stream>>>(key, Xb, nx4);
    cvt_f32_bf16<<<(nw4 + 255) / 256, 256, 0, stream>>>(w_k, Wbuf, nw4);
    gemm_bf16<1, 1><<<gg, 256, 0, stream>>>(Xb, Wbuf, b_k, Ktbuf);
    // V projection (per-head transposed)
    cvt_f32_bf16<<<(nx4 + 255) / 256, 256, 0, stream>>>(value, Xb, nx4);
    cvt_f32_bf16<<<(nw4 + 255) / 256, 256, 0, stream>>>(w_v, Wbuf, nw4);
    gemm_bf16<0, 1><<<gg, 256, 0, stream>>>(Xb, Wbuf, b_v, Vtbuf);

    // attention core
    const int nkv = BATCH * NHEADS * DK * DK;  // 262144
    zero_f32<<<(nkv + 255) / 256, 256, 0, stream>>>(KVf, nkv);
    kv_accum<<<dim3(8, BATCH * NHEADS), 128, 0, stream>>>(Ktbuf, Vtbuf, KVf);
    ksum_kernel<<<(BATCH * NHEADS * DK) / 8, 256, 0, stream>>>(Ktbuf, Ksumb);
    {
        const int n4 = nkv / 4;
        cvt_f32_bf16<<<(n4 + 255) / 256, 256, 0, stream>>>(KVf, KVtb, n4);
    }
    attn_apply<<<dim3(SEQ / 64, BATCH * NHEADS), 128, 0, stream>>>(Qbuf, KVtb, Ksumb, Zbuf);

    // output projection (f32 out)
    cvt_f32_bf16<<<(nw4 + 255) / 256, 256, 0, stream>>>(w_o, Wbuf, nw4);
    gemm_bf16<0, 2><<<gg, 256, 0, stream>>>(Zbuf, Wbuf, b_o, (float*)d_out);
}
